// Bert_66803921322538
// MI455X (gfx1250) — matrix operand order repaired and checked by reading, not compiled, not run on hardware
//
#include <hip/hip_runtime.h>
#include <hip/hip_bf16.h>

// ---------------- constants ----------------
constexpr int kS  = 512;
constexpr int kB  = 16;
constexpr int kH  = 768;
constexpr int kNH = 12;
constexpr int kHD = 64;
constexpr float kEps   = 1e-7f;
constexpr float kScale = 0.07216878364870323f; // 1/sqrt(3*64)

typedef __attribute__((ext_vector_type(16))) _Float16 v16h;
typedef __attribute__((ext_vector_type(8)))  _Float16 v8h;
typedef __attribute__((ext_vector_type(8)))  float    v8f;

// ---------------- WMMA fragment helpers (ISA 7.12.2 layouts, wave32) ----------------
__device__ __forceinline__ v16h load_a_frag(const _Float16* __restrict__ A, int lda, int m0, int k0) {
  int lane = threadIdx.x & 31;
  const _Float16* base = A + (size_t)(m0 + (lane & 15)) * lda + (k0 + ((lane >> 4) << 3));
  v8h lo = *(const v8h*)(base);
  v8h hi = *(const v8h*)(base + 16);
  v16h a;
#pragma unroll
  for (int i = 0; i < 8; ++i) { a[i] = lo[i]; a[i + 8] = hi[i]; }
  return a;
}

__device__ __forceinline__ v16h load_b_frag(const _Float16* __restrict__ Bt, int ldb, int n0, int k0) {
  int lane = threadIdx.x & 31;
  const _Float16* base = Bt + (size_t)(n0 + (lane & 15)) * ldb + (k0 + ((lane >> 4) << 3));
  v8h lo = *(const v8h*)(base);
  v8h hi = *(const v8h*)(base + 16);
  v16h b;
#pragma unroll
  for (int i = 0; i < 8; ++i) { b[i] = lo[i]; b[i + 8] = hi[i]; }
  return b;
}

// 32Mx64N register-blocked GEMM: acc[2][4] of 16x16 f32 tiles; A reused over 4 N-tiles,
// B reused over 2 M-tiles -> 8 wmma per 12 b128 loads per K-step.
__device__ __forceinline__ void wmma_block(const _Float16* __restrict__ A, int lda,
                                           const _Float16* __restrict__ Bt, int ldb,
                                           int m0, int n0, int K, v8f (&acc)[2][4]) {
  for (int k0 = 0; k0 < K; k0 += 32) {
    v16h a0 = load_a_frag(A, lda, m0, k0);
    v16h a1 = load_a_frag(A, lda, m0 + 16, k0);
#pragma unroll
    for (int nb = 0; nb < 4; ++nb) {
      v16h b = load_b_frag(Bt, ldb, n0 + nb * 16, k0);
      acc[0][nb] = __builtin_amdgcn_wmma_f32_16x16x32_f16(false, a0, false, b, (short)0, acc[0][nb], false, false);
      acc[1][nb] = __builtin_amdgcn_wmma_f32_16x16x32_f16(false, a1, false, b, (short)0, acc[1][nb], false, false);
    }
  }
}

// ---------------- reductions ----------------
__device__ __forceinline__ float wave_sum(float v) {
#pragma unroll
  for (int o = 16; o > 0; o >>= 1) v += __shfl_xor(v, o, 32);
  return v;
}
__device__ __forceinline__ float wave_max(float v) {
#pragma unroll
  for (int o = 16; o > 0; o >>= 1) v = fmaxf(v, __shfl_xor(v, o, 32));
  return v;
}

// ---------------- small prep kernels ----------------
__global__ void k_cvt_f16(const float* __restrict__ src, _Float16* __restrict__ dst, int n) {
  int i = blockIdx.x * blockDim.x + threadIdx.x;
  if (i < n) dst[i] = (_Float16)src[i];
}

__global__ void k_relpad(const float* __restrict__ src, _Float16* __restrict__ dst) {
  int i = blockIdx.x * blockDim.x + threadIdx.x; // 64*768 total, pad row 63 with zeros
  if (i < 63 * kH) dst[i] = (_Float16)src[i];
  else if (i < 64 * kH) dst[i] = (_Float16)0.0f;
}

// position_indices (512x512 int, values 0..62) -> bytes for LDS staging
__global__ void k_pos8(const int* __restrict__ pos, unsigned char* __restrict__ p8) {
  int i = blockIdx.x * blockDim.x + threadIdx.x;
  if (i < kS * kS) p8[i] = (unsigned char)pos[i];
}

// LN (no affine) of hidden_states rows (8192 x 768) -> f16
__global__ void k_ln_f16(const float* __restrict__ x, _Float16* __restrict__ o) {
  int row = blockIdx.x;
  const float* xr = x + (size_t)row * kH;
  int t = threadIdx.x;
  float v0 = xr[t], v1 = xr[t + 256], v2 = xr[t + 512];
  float s = v0 + v1 + v2;
  float s2 = v0 * v0 + v1 * v1 + v2 * v2;
  __shared__ float rs[8], rq[8];
  s = wave_sum(s); s2 = wave_sum(s2);
  int wid = t >> 5;
  if ((t & 31) == 0) { rs[wid] = s; rq[wid] = s2; }
  __syncthreads();
  if (t == 0) {
    float a = 0.f, b = 0.f;
#pragma unroll
    for (int i = 0; i < 8; ++i) { a += rs[i]; b += rq[i]; }
    rs[0] = a; rq[0] = b;
  }
  __syncthreads();
  float mu  = rs[0] * (1.0f / kH);
  float var = rq[0] * (1.0f / kH) - mu * mu;
  float inv = rsqrtf(var + kEps);
  _Float16* orow = o + (size_t)row * kH;
  orow[t]       = (_Float16)((v0 - mu) * inv);
  orow[t + 256] = (_Float16)((v1 - mu) * inv);
  orow[t + 512] = (_Float16)((v2 - mu) * inv);
}

// final LN with gamma/beta -> f32 output
__global__ void k_ln_out(const float* __restrict__ x, const float* __restrict__ g,
                         const float* __restrict__ be, float* __restrict__ o) {
  int row = blockIdx.x;
  const float* xr = x + (size_t)row * kH;
  int t = threadIdx.x;
  float v0 = xr[t], v1 = xr[t + 256], v2 = xr[t + 512];
  float s = v0 + v1 + v2;
  float s2 = v0 * v0 + v1 * v1 + v2 * v2;
  __shared__ float rs[8], rq[8];
  s = wave_sum(s); s2 = wave_sum(s2);
  int wid = t >> 5;
  if ((t & 31) == 0) { rs[wid] = s; rq[wid] = s2; }
  __syncthreads();
  if (t == 0) {
    float a = 0.f, b = 0.f;
#pragma unroll
    for (int i = 0; i < 8; ++i) { a += rs[i]; b += rq[i]; }
    rs[0] = a; rq[0] = b;
  }
  __syncthreads();
  float mu  = rs[0] * (1.0f / kH);
  float var = rq[0] * (1.0f / kH) - mu * mu;
  float inv = rsqrtf(var + kEps);
  float* orow = o + (size_t)row * kH;
  orow[t]       = (v0 - mu) * inv * g[t]       + be[t];
  orow[t + 256] = (v1 - mu) * inv * g[t + 256] + be[t + 256];
  orow[t + 512] = (v2 - mu) * inv * g[t + 512] + be[t + 512];
}

// ---------------- GEMM kernels (32x64 block / wave, 8 waves / block) ----------------
// qk = hs @ w_qk^T + b_qk ; split into q,k stored (B,NH,S,HD) f16
__global__ void k_gemm_qk(const _Float16* __restrict__ hs, const _Float16* __restrict__ wqk,
                          const float* __restrict__ bqk,
                          _Float16* __restrict__ qh, _Float16* __restrict__ kh) {
  int t = blockIdx.x * 8 + (threadIdx.x >> 5); // 6144 blocks32x64: 256 x 24
  int m0 = (t / 24) * 32, n0 = (t % 24) * 64;
  v8f acc[2][4] = {};
  wmma_block(hs, kH, wqk, kH, m0, n0, kH, acc);
  int lane = threadIdx.x & 31;
#pragma unroll
  for (int nb = 0; nb < 4; ++nb) {
    int n = n0 + nb * 16 + (lane & 15);
    float bias = bqk[n];
    int n2 = (n < kH) ? n : n - kH;
    int h = n2 >> 6, d = n2 & 63;
    _Float16* dst = (n < kH) ? qh : kh;
#pragma unroll
    for (int mb = 0; mb < 2; ++mb) {
      int mbase = m0 + mb * 16 + ((lane >> 4) << 3);
#pragma unroll
      for (int j = 0; j < 8; ++j) {
        int m = mbase + j;
        int s = m >> 4, bidx = m & 15; // r = s*16 + b
        dst[(((size_t)bidx * kNH + h) * kS + s) * kHD + d] = (_Float16)(acc[mb][nb][j] + bias);
      }
    }
  }
}

// v = hs @ w_v^T + b_v ; stored TRANSPOSED (B,NH,HD,S)
__global__ void k_gemm_v(const _Float16* __restrict__ hs, const _Float16* __restrict__ wv,
                         const float* __restrict__ bv, _Float16* __restrict__ vT) {
  int t = blockIdx.x * 8 + (threadIdx.x >> 5); // 3072 blocks32x64: 256 x 12
  int m0 = (t / 12) * 32, n0 = (t % 12) * 64;
  v8f acc[2][4] = {};
  wmma_block(hs, kH, wv, kH, m0, n0, kH, acc);
  int lane = threadIdx.x & 31;
#pragma unroll
  for (int nb = 0; nb < 4; ++nb) {
    int n = n0 + nb * 16 + (lane & 15);
    float bias = bv[n];
    int h = n >> 6, d = n & 63;
#pragma unroll
    for (int mb = 0; mb < 2; ++mb) {
      int mbase = m0 + mb * 16 + ((lane >> 4) << 3);
#pragma unroll
      for (int j = 0; j < 8; ++j) {
        int m = mbase + j;
        int s = m >> 4, bidx = m & 15;
        vT[(((size_t)bidx * kNH + h) * kHD + d) * kS + s] = (_Float16)(acc[mb][nb][j] + bias);
      }
    }
  }
}

// rel = relpad(64x768) @ w_qk^T + b_qk ; qp/kp stored (NH, 64pos, 64d)
__global__ void k_gemm_rel(const _Float16* __restrict__ relpad, const _Float16* __restrict__ wqk,
                           const float* __restrict__ bqk,
                           _Float16* __restrict__ qp, _Float16* __restrict__ kp) {
  int t = blockIdx.x * 8 + (threadIdx.x >> 5); // 48 blocks32x64: 2 x 24
  int m0 = (t / 24) * 32, n0 = (t % 24) * 64;
  v8f acc[2][4] = {};
  wmma_block(relpad, kH, wqk, kH, m0, n0, kH, acc);
  int lane = threadIdx.x & 31;
#pragma unroll
  for (int nb = 0; nb < 4; ++nb) {
    int n = n0 + nb * 16 + (lane & 15);
    float bias = bqk[n];
    int n2 = (n < kH) ? n : n - kH;
    int h = n2 >> 6, d = n2 & 63;
    _Float16* dst = (n < kH) ? qp : kp;
#pragma unroll
    for (int mb = 0; mb < 2; ++mb) {
      int mbase = m0 + mb * 16 + ((lane >> 4) << 3);
#pragma unroll
      for (int j = 0; j < 8; ++j) {
        int p = mbase + j; // p==63 junk, never gathered
        dst[((size_t)h * 64 + p) * 64 + d] = (_Float16)(acc[mb][nb][j] + bias);
      }
    }
  }
}

// sqp[bh,q,p] = q_bh(512x64) @ kp_h(64x64)^T
__global__ void k_sqp(const _Float16* __restrict__ qh, const _Float16* __restrict__ kp,
                      float* __restrict__ sqp) {
  int b = blockIdx.z, h = blockIdx.y, bh = b * kNH + h;
  int t = blockIdx.x * 8 + (threadIdx.x >> 5); // 16 tiles: m strips of 32, N=64
  int m0 = t * 32;
  const _Float16* A  = qh + (size_t)bh * kS * kHD;
  const _Float16* Bt = kp + (size_t)h * 64 * 64;
  v8f acc[2][4] = {};
  wmma_block(A, kHD, Bt, 64, m0, 0, kHD, acc);
  int lane = threadIdx.x & 31;
  float* out = sqp + (size_t)bh * kS * 64;
#pragma unroll
  for (int nb = 0; nb < 4; ++nb) {
    int n = nb * 16 + (lane & 15);
#pragma unroll
    for (int mb = 0; mb < 2; ++mb) {
      int mbase = m0 + mb * 16 + ((lane >> 4) << 3);
#pragma unroll
      for (int j = 0; j < 8; ++j) out[(size_t)(mbase + j) * 64 + n] = acc[mb][nb][j];
    }
  }
}

// spk[bh,p,k] = qp_h(64x64) @ k_bh(512x64)^T
__global__ void k_spk(const _Float16* __restrict__ qp, const _Float16* __restrict__ kh,
                      float* __restrict__ spk) {
  int b = blockIdx.z, h = blockIdx.y, bh = b * kNH + h;
  int t = blockIdx.x * 8 + (threadIdx.x >> 5); // 16 tiles: 2 x 8
  int m0 = (t >> 3) * 32, n0 = (t & 7) * 64;
  const _Float16* A  = qp + (size_t)h * 64 * 64;
  const _Float16* Bt = kh + (size_t)bh * kS * kHD;
  v8f acc[2][4] = {};
  wmma_block(A, 64, Bt, kHD, m0, n0, kHD, acc);
  int lane = threadIdx.x & 31;
  float* out = spk + (size_t)bh * 64 * kS;
#pragma unroll
  for (int nb = 0; nb < 4; ++nb) {
    int n = n0 + nb * 16 + (lane & 15);
#pragma unroll
    for (int mb = 0; mb < 2; ++mb) {
      int mbase = m0 + mb * 16 + ((lane >> 4) << 3);
#pragma unroll
      for (int j = 0; j < 8; ++j) out[(size_t)(mbase + j) * kS + n] = acc[mb][nb][j];
    }
  }
}

// scores[bh,q,k] = (q.k^T + sqp[q,pos] + spk[pos,k]) * scale
// One block = one 32-row q-strip x all 512 keys (wave w covers cols w*64..w*64+63).
// The strip's position indices (32x512 bytes) are staged in LDS via async global->LDS.
__global__ void k_scores(const _Float16* __restrict__ qh, const _Float16* __restrict__ kh,
                         const float* __restrict__ sqp, const float* __restrict__ spk,
                         const unsigned char* __restrict__ pos8, float* __restrict__ scores) {
  __shared__ __align__(16) unsigned char lpos[32 * kS]; // 16 KB
  int b = blockIdx.z, h = blockIdx.y, bh = b * kNH + h;
  int qs = blockIdx.x * 32;

  // async-copy the 16KB index strip into LDS (ASYNCcnt path)
  {
    int tt = threadIdx.x; // 256 threads x 4 x 16B = 16KB
    const unsigned char* g = pos8 + (size_t)qs * kS;
    unsigned lbase = (unsigned)(size_t)(&lpos[0]);
#pragma unroll
    for (int i = 0; i < 4; ++i) {
      unsigned loff = lbase + (unsigned)(tt * 16 + i * 4096);
      unsigned long long ga = (unsigned long long)(size_t)(g + tt * 16 + i * 4096);
      asm volatile("global_load_async_to_lds_b128 %0, %1, off" :: "v"(loff), "v"(ga) : "memory");
    }
    asm volatile("s_wait_asynccnt 0" ::: "memory");
  }
  __syncthreads();

  int w = threadIdx.x >> 5;
  int n0 = w * 64;
  const _Float16* A  = qh + (size_t)bh * kS * kHD;
  const _Float16* Bt = kh + (size_t)bh * kS * kHD;
  v8f acc[2][4] = {};
  wmma_block(A, kHD, Bt, kHD, qs, n0, kHD, acc);

  int lane = threadIdx.x & 31;
  const float* sq = sqp + (size_t)bh * kS * 64;
  const float* sp = spk + (size_t)bh * 64 * kS;
  float* srow = scores + (size_t)bh * kS * kS;
#pragma unroll
  for (int nb = 0; nb < 4; ++nb) {
    int kk = n0 + nb * 16 + (lane & 15);
#pragma unroll
    for (int mb = 0; mb < 2; ++mb) {
      int mbase = qs + mb * 16 + ((lane >> 4) << 3);
#pragma unroll
      for (int j = 0; j < 8; ++j) {
        int q = mbase + j;
        int p = (int)lpos[(q - qs) * kS + kk];
        srow[(size_t)q * kS + kk] =
            (acc[mb][nb][j] + sq[(size_t)q * 64 + p] + sp[(size_t)p * kS + kk]) * kScale;
      }
    }
  }
}

// returned_scores = mean over heads (pre-mask)
__global__ void k_mean(const float* __restrict__ scores, float* __restrict__ out2) {
  size_t i = (size_t)blockIdx.x * blockDim.x + threadIdx.x; // B*S*S = 4194304
  int b = (int)(i >> 18);
  size_t r = i & 262143;
  float s = 0.f;
#pragma unroll
  for (int h = 0; h < kNH; ++h) s += scores[(((size_t)(b * kNH + h)) << 18) + r];
  out2[i] = s * (1.0f / kNH);
}

// masked softmax per row; write f16 probs in place over the f32 row (first half)
__global__ void k_softmax(float* __restrict__ scores, const int* __restrict__ mask) {
  int row = blockIdx.x; // bh*512 + q
  int bh = row >> 9;
  int b = bh / kNH;
  float* sr = scores + (size_t)row * kS;
  int t = threadIdx.x; // 512 threads, 16 waves
  float xv = sr[t];
  int msk = mask[b * kS + t];
  float x = msk ? -3.4e38f : xv;
  __shared__ float red[16];
  float wm = wave_max(x);
  int wid = t >> 5;
  if ((t & 31) == 0) red[wid] = wm;
  __syncthreads();
  if (t == 0) {
    float m = red[0];
#pragma unroll
    for (int i = 1; i < 16; ++i) m = fmaxf(m, red[i]);
    red[0] = m;
  }
  __syncthreads();
  float rowmax = red[0];
  float e = msk ? 0.0f : __expf(x - rowmax);
  __syncthreads();
  float ws = wave_sum(e);
  if ((t & 31) == 0) red[wid] = ws;
  __syncthreads();
  if (t == 0) {
    float s = 0.f;
#pragma unroll
    for (int i = 0; i < 16; ++i) s += red[i];
    red[0] = s;
  }
  __syncthreads();
  float p = e / red[0];
  __syncthreads(); // all f32 reads complete before aliased f16 writes
  ((_Float16*)sr)[t] = (_Float16)p;
}

// ctx = probs(512x512 f16, lda=1024) @ vT(64x512)^T ; write ctx as (S,B,H) f16
__global__ void k_ctx(const float* __restrict__ scores, const _Float16* __restrict__ vT,
                      _Float16* __restrict__ ctx) {
  int b = blockIdx.z, h = blockIdx.y, bh = b * kNH + h;
  int t = blockIdx.x * 8 + (threadIdx.x >> 5); // 16 tiles: m strips of 32, N=64
  int m0 = t * 32;
  const _Float16* A  = (const _Float16*)(scores + (size_t)bh * kS * kS); // f16 rows, stride 1024
  const _Float16* Bt = vT + (size_t)bh * kHD * kS;
  v8f acc[2][4] = {};
  wmma_block(A, 1024, Bt, kS, m0, 0, kS, acc);
  int lane = threadIdx.x & 31;
#pragma unroll
  for (int nb = 0; nb < 4; ++nb) {
    int d = nb * 16 + (lane & 15);
#pragma unroll
    for (int mb = 0; mb < 2; ++mb) {
      int mbase = m0 + mb * 16 + ((lane >> 4) << 3);
#pragma unroll
      for (int j = 0; j < 8; ++j) {
        int s = mbase + j;
        ctx[((size_t)s * kB + b) * kH + h * kHD + d] = (_Float16)acc[mb][nb][j];
      }
    }
  }
}

// out_pre = ctx @ w_o^T + b_o  (f32)
__global__ void k_gemm_out(const _Float16* __restrict__ ctx, const _Float16* __restrict__ wo,
                           const float* __restrict__ bo, float* __restrict__ outpre) {
  int t = blockIdx.x * 8 + (threadIdx.x >> 5); // 3072 blocks32x64: 256 x 12
  int m0 = (t / 12) * 32, n0 = (t % 12) * 64;
  v8f acc[2][4] = {};
  wmma_block(ctx, kH, wo, kH, m0, n0, kH, acc);
  int lane = threadIdx.x & 31;
#pragma unroll
  for (int nb = 0; nb < 4; ++nb) {
    int n = n0 + nb * 16 + (lane & 15);
    float bias = bo[n];
#pragma unroll
    for (int mb = 0; mb < 2; ++mb) {
      int mbase = m0 + mb * 16 + ((lane >> 4) << 3);
#pragma unroll
      for (int j = 0; j < 8; ++j)
        outpre[(size_t)(mbase + j) * kH + n] = acc[mb][nb][j] + bias;
    }
  }
}

// ---------------- host ----------------
extern "C" void kernel_launch(void* const* d_in, const int* in_sizes, int n_in,
                              void* d_out, int out_size, void* d_ws, size_t ws_size,
                              hipStream_t stream) {
  (void)in_sizes; (void)n_in; (void)out_size; (void)ws_size;
  const float* hidden = (const float*)d_in[0];
  const float* relemb = (const float*)d_in[1];
  const float* w_qk   = (const float*)d_in[2];
  const float* b_qk   = (const float*)d_in[3];
  const float* w_v    = (const float*)d_in[4];
  const float* b_v    = (const float*)d_in[5];
  const float* w_o    = (const float*)d_in[6];
  const float* b_o    = (const float*)d_in[7];
  const float* gamma  = (const float*)d_in[8];
  const float* beta   = (const float*)d_in[9];
  const int*   mask   = (const int*)d_in[10];
  const int*   pos    = (const int*)d_in[11];

  float* out1 = (float*)d_out;               // (S,B,H)
  float* out2 = out1 + (size_t)kS * kB * kH; // (B,S,S)

  char* ws = (char*)d_ws;
  size_t off = 0;
  auto alloc = [&](size_t bytes) -> char* {
    off = (off + 255) & ~(size_t)255;
    char* p = ws + off;
    off += bytes;
    return p;
  };

  const size_t rowsMB = (size_t)kS * kB; // 8192
  _Float16* wqk_h = (_Float16*)alloc((size_t)2 * kH * kH * 2);
  _Float16* wv_h  = (_Float16*)alloc((size_t)kH * kH * 2);
  _Float16* wo_h  = (_Float16*)alloc((size_t)kH * kH * 2);
  _Float16* rel_h = (_Float16*)alloc((size_t)64 * kH * 2);
  _Float16* hs_h  = (_Float16*)alloc(rowsMB * kH * 2);
  _Float16* q_h   = (_Float16*)alloc((size_t)kB * kNH * kS * kHD * 2);
  _Float16* k_h   = (_Float16*)alloc((size_t)kB * kNH * kS * kHD * 2);
  _Float16* vT_h  = (_Float16*)alloc((size_t)kB * kNH * kHD * kS * 2);
  _Float16* qp_h  = (_Float16*)alloc((size_t)kNH * 64 * 64 * 2);
  _Float16* kp_h  = (_Float16*)alloc((size_t)kNH * 64 * 64 * 2);
  unsigned char* pos8 = (unsigned char*)alloc((size_t)kS * kS);
  float*    sqp   = (float*)alloc((size_t)kB * kNH * kS * 64 * 4);
  float*    spk   = (float*)alloc((size_t)kB * kNH * 64 * kS * 4);
  float*    scor  = (float*)alloc((size_t)kB * kNH * kS * kS * 4);
  _Float16* ctx_h = (_Float16*)alloc(rowsMB * kH * 2);
  float*    opre  = (float*)alloc(rowsMB * kH * 4);

  // 1) weight / embedding / index conversion
  {
    int n = 2 * kH * kH;
    k_cvt_f16<<<(n + 255) / 256, 256, 0, stream>>>(w_qk, wqk_h, n);
    n = kH * kH;
    k_cvt_f16<<<(n + 255) / 256, 256, 0, stream>>>(w_v, wv_h, n);
    k_cvt_f16<<<(n + 255) / 256, 256, 0, stream>>>(w_o, wo_h, n);
    k_relpad<<<(64 * kH + 255) / 256, 256, 0, stream>>>(relemb, rel_h);
    k_pos8<<<(kS * kS + 255) / 256, 256, 0, stream>>>(pos, pos8);
  }
  // 2) input LayerNorm -> f16
  k_ln_f16<<<(int)rowsMB, 256, 0, stream>>>(hidden, hs_h);
  // 3) projections
  k_gemm_qk<<<768, 256, 0, stream>>>(hs_h, wqk_h, b_qk, q_h, k_h);
  k_gemm_v<<<384, 256, 0, stream>>>(hs_h, wv_h, b_v, vT_h);
  k_gemm_rel<<<6, 256, 0, stream>>>(rel_h, wqk_h, b_qk, qp_h, kp_h);
  // 4) positional score matrices
  dim3 gbh2(2, kNH, kB);
  k_sqp<<<gbh2, 256, 0, stream>>>(q_h, kp_h, sqp);
  k_spk<<<gbh2, 256, 0, stream>>>(qp_h, k_h, spk);
  // 5) attention scores + gathers (async LDS staging of indices)
  k_scores<<<dim3(16, kNH, kB), 256, 0, stream>>>(q_h, k_h, sqp, spk, pos8, scor);
  // 6) head-mean (pre-mask) -> second output
  k_mean<<<16384, 256, 0, stream>>>(scor, out2);
  // 7) masked softmax -> f16 probs in place
  k_softmax<<<kB * kNH * kS, 512, 0, stream>>>(scor, mask);
  // 8) context
  k_ctx<<<gbh2, 256, 0, stream>>>(scor, vT_h, ctx_h);
  // 9) output projection + final LN
  k_gemm_out<<<384, 256, 0, stream>>>(ctx_h, wo_h, b_o, opre);
  k_ln_out<<<(int)rowsMB, 256, 0, stream>>>(opre, gamma, beta, out1);
}
